// FiLMConditionedCNN_19602230739540
// MI455X (gfx1250) — matrix operand order repaired and checked by reading, not compiled, not run on hardware
//
#include <hip/hip_runtime.h>

// ---------------------------------------------------------------------------
// FiLM-conditioned CNN on MI455X (gfx1250).
// - All GEMM-like ops on the bf16 WMMA pipe (v_wmma_f32_16x16x32_bf16),
//   2 M-subtiles per wave (dual accumulators) to hide XDL latency and halve
//   LDS traffic per WMMA.
// - Activations NHWC with channel padding -> implicit-GEMM A fragments are
//   contiguous 16B global loads.
// - Weights packed as B^T [Npad][Kpad]; per-block N-tile staged to LDS with
//   CDNA5 async loads (global_load_async_to_lds_b128 + s_wait_asynccnt).
// ---------------------------------------------------------------------------

typedef __attribute__((ext_vector_type(16))) __bf16 v16bf;
typedef __attribute__((ext_vector_type(8)))  float  v8f;

union Frag16 {
    v16bf v;
    uint4 q[2];
    unsigned short u[16];
};

__device__ __forceinline__ unsigned short f2bf(float f) {
    union { float f; unsigned u; } x; x.f = f;
    unsigned r = x.u + 0x7FFFu + ((x.u >> 16) & 1u);   // round-to-nearest-even
    return (unsigned short)(r >> 16);
}
__device__ __forceinline__ float bf2f(unsigned short b) {
    union { unsigned u; float f; } x; x.u = ((unsigned)b) << 16;
    return x.f;
}

static constexpr int BATCH = 4096;
static constexpr int EMB   = 56;
static constexpr int CONDP = 128;   // 113 real cond features, padded to 128

// Stage a contiguous bf16 tile (global -> LDS) with CDNA5 async LDS loads.
// nElems must be a multiple of 8 (16 bytes). Caller must __syncthreads after.
__device__ __forceinline__ void stage_tile_async(const unsigned short* __restrict__ g,
                                                 unsigned short* lds, int nElems, int tid) {
    for (int i = tid * 8; i < nElems; i += 128 * 8) {
        unsigned laddr = (unsigned)(size_t)(lds + i);   // flat addr[31:0] == LDS offset
        const unsigned short* gaddr = g + i;
        asm volatile("global_load_async_to_lds_b128 %0, %1, off"
                     :: "v"(laddr), "v"(gaddr) : "memory");
    }
    asm volatile("s_wait_asynccnt 0x0" ::: "memory");
}

// ---------------------------------------------------------------------------
// Prep kernels
// ---------------------------------------------------------------------------

// images fp32 NCHW [B][10][32][32] -> bf16 NHWC padded [B][32][32][16]
__global__ void img_to_nhwc(const float* __restrict__ in,
                            unsigned short* __restrict__ out) {
    int idx = blockIdx.x * blockDim.x + threadIdx.x;
    if (idx >= BATCH * 32 * 32 * 16) return;
    int c = idx & 15, p = idx >> 4;          // p = b*1024 + y*32 + x
    int b = p >> 10, yx = p & 1023;
    float v = (c < 10) ? in[((b * 10 + c) << 10) + yx] : 0.f;
    out[idx] = f2bf(v);
}

// cond[b][0:56]=embed[z[2b]], [56:112]=embed[z[2b+1]], [112]=|p0-p1|, rest 0
__global__ void build_cond(const int* __restrict__ z,
                           const float* __restrict__ pos,
                           const float* __restrict__ embed,
                           unsigned short* __restrict__ cond) {
    int idx = blockIdx.x * blockDim.x + threadIdx.x;
    if (idx >= BATCH * CONDP) return;
    int b = idx >> 7, j = idx & 127;
    float v = 0.f;
    if (j < EMB) {
        v = embed[z[2 * b] * EMB + j];
    } else if (j < 2 * EMB) {
        v = embed[z[2 * b + 1] * EMB + (j - EMB)];
    } else if (j == 112) {
        float dx = pos[(2 * b) * 3 + 0] - pos[(2 * b + 1) * 3 + 0];
        float dy = pos[(2 * b) * 3 + 1] - pos[(2 * b + 1) * 3 + 1];
        float dz = pos[(2 * b) * 3 + 2] - pos[(2 * b + 1) * 3 + 2];
        v = sqrtf(dx * dx + dy * dy + dz * dz);
    }
    cond[idx] = f2bf(v);
}

// Pack fp32 FC weight W[N][K] into bf16 B^T layout Bw[NP][KP] (zero padded)
__global__ void pack_fc(const float* __restrict__ W,
                        unsigned short* __restrict__ Bw,
                        int N, int K, int NP, int KP) {
    int idx = blockIdx.x * blockDim.x + threadIdx.x;
    if (idx >= NP * KP) return;
    int n = idx / KP, k = idx - n * KP;
    Bw[idx] = f2bf((n < N && k < K) ? W[n * K + k] : 0.f);
}

// Pack conv weight [COUT][CINr][3][3] into Bw[NP][KP], K = (ky*3+kx)*CPIN + ci
__global__ void pack_conv(const float* __restrict__ W,
                          unsigned short* __restrict__ Bw,
                          int COUT, int CINr, int CPIN, int NP, int KP) {
    int idx = blockIdx.x * blockDim.x + threadIdx.x;
    if (idx >= NP * KP) return;
    int n = idx / KP, k = idx - n * KP;
    float v = 0.f;
    if (n < COUT && k < 9 * CPIN) {
        int seg = k / CPIN, ci = k - seg * CPIN;
        if (ci < CINr) {
            int ky = seg / 3, kx = seg - ky * 3;
            v = W[((n * CINr + ci) * 3 + ky) * 3 + kx];
        }
    }
    Bw[idx] = f2bf(v);
}

// head_w1 [256][208] where orig k = co*4 + (oy*2+ox); our act4 flat index is
// k' = (oy*2+ox)*64 + co  (NHWC, CPOUT=64). Pack into Bw[256][256].
__global__ void pack_head1(const float* __restrict__ W,
                           unsigned short* __restrict__ Bw) {
    int idx = blockIdx.x * blockDim.x + threadIdx.x;
    if (idx >= 256 * 256) return;
    int n = idx >> 8, kp = idx & 255;
    int pos = kp >> 6, co = kp & 63;
    float v = (co < 52) ? W[n * 208 + co * 4 + pos] : 0.f;
    Bw[idx] = f2bf(v);
}

// Fold BN (eval) + conv bias: y = sc*conv + sh
__global__ void fuse_bn(const float* __restrict__ bnw, const float* __restrict__ bnb,
                        const float* __restrict__ mean, const float* __restrict__ var,
                        const float* __restrict__ convb,
                        float* __restrict__ scale, float* __restrict__ shift, int C) {
    int c = blockIdx.x * blockDim.x + threadIdx.x;
    if (c >= C) return;
    float sc = bnw[c] * rsqrtf(var[c] + 1e-5f);
    scale[c] = sc;
    shift[c] = bnb[c] + sc * (convb[c] - mean[c]);
}

// ---------------------------------------------------------------------------
// Dense-A WMMA GEMM: out[M][ldo] = A[M][KP] @ Bt[NP][KP]^T (+bias, relu)
// Block = 128 threads (4 waves) covering 128 rows: wave w owns M-subtiles at
// blockIdx.x*128 + w*16 and +64 (dual accumulators share the B fragment).
// grid = (M/128, NP/16). The block's 16-row slice of Bt is contiguous ->
// async-copied to LDS.
// ---------------------------------------------------------------------------
template <int KP, bool RELU, bool OUT_BF16>
__global__ void gemm_wmma(const unsigned short* __restrict__ A,
                          const unsigned short* __restrict__ Bt,   // [NP][KP]
                          const float* __restrict__ bias,
                          void* __restrict__ out,
                          int M, int N, int ldo) {
    constexpr int KSTEPS = KP / 32;
    __shared__ __align__(16) unsigned short ldsB[16 * KP];

    const int ntile = blockIdx.y;
    const int tid   = threadIdx.x;
    stage_tile_async(Bt + (size_t)ntile * 16 * KP, ldsB, 16 * KP, tid);
    __syncthreads();

    const int wave = tid >> 5, lane = tid & 31;
    const int half = lane >> 4, lm = lane & 15;
    const int mbase0 = blockIdx.x * 128 + wave * 16;   // second tile at +64
    const unsigned short* arow = A + (size_t)(mbase0 + lm) * KP;

    v8f acc0 = {}, acc1 = {};
#pragma unroll
    for (int ks = 0; ks < KSTEPS; ++ks) {
        const int k0 = ks * 32;
        Frag16 a0, a1, b;
        // A (16-bit 16x32): VGPR0-3 <-> K = half*8+0..7, VGPR4-7 <-> 16+half*8+0..7
        a0.q[0] = *reinterpret_cast<const uint4*>(arow + k0 + half * 8);
        a0.q[1] = *reinterpret_cast<const uint4*>(arow + k0 + 16 + half * 8);
        a1.q[0] = *reinterpret_cast<const uint4*>(arow + 64 * KP + k0 + half * 8);
        a1.q[1] = *reinterpret_cast<const uint4*>(arow + 64 * KP + k0 + 16 + half * 8);
        // B (32x16): same K order as A: K = half*8+0..7, 16+half*8+0..7
        const unsigned short* bp = &ldsB[lm * KP + k0 + half * 8];
        b.q[0] = *reinterpret_cast<const uint4*>(bp);
        b.q[1] = *reinterpret_cast<const uint4*>(bp + 16);
        acc0 = __builtin_amdgcn_wmma_f32_16x16x32_bf16(
            false, a0.v, false, b.v, (short)0, acc0, false, false);
        acc1 = __builtin_amdgcn_wmma_f32_16x16x32_bf16(
            false, a1.v, false, b.v, (short)0, acc1, false, false);
    }

    const int n = ntile * 16 + lm;          // C/D: N = lane%16, M = vgpr + 8*half
    if (n < N) {
        const float bs = bias[n];
#pragma unroll
        for (int t = 0; t < 2; ++t) {
            const v8f& acc = t ? acc1 : acc0;
            const int mb = mbase0 + t * 64;
#pragma unroll
            for (int r = 0; r < 8; ++r) {
                int gm = mb + r + 8 * half;
                float y = acc[r] + bs;
                if (RELU) y = fmaxf(y, 0.f);
                if (OUT_BF16)
                    ((unsigned short*)out)[(size_t)gm * ldo + n] = f2bf(y);
                else
                    ((float*)out)[(size_t)gm * ldo + n] = y;
            }
        }
    }
}

// ---------------------------------------------------------------------------
// Implicit-GEMM conv (3x3, stride 2, pad 1) + BN + FiLM + ReLU.
// NHWC bf16 in [B][IH][IH][CPIN], out [B][OH][OH][CPOUT] (pad channels = 0).
// K = (ky*3+kx)*CPIN + ci, zero-padded to KP. Each lane's 8-element A chunks
// are contiguous 16B loads (CPIN multiple of 8 => no segment straddle).
// Dual M-subtiles per wave (rows m and m+64), block covers 128 rows.
// ---------------------------------------------------------------------------
template <int CINR, int CPIN, int COUT, int CPOUT, int IH, int OH>
__global__ void conv_film_wmma(const unsigned short* __restrict__ actin,
                               const unsigned short* __restrict__ Wt,  // [NP][KP]
                               const float* __restrict__ scale,
                               const float* __restrict__ shift,
                               const float* __restrict__ gamma,        // [B][COUT]
                               const float* __restrict__ beta,         // [B][COUT]
                               unsigned short* __restrict__ actout) {
    constexpr int KREAL  = 9 * CPIN;
    constexpr int KP     = (KREAL + 31) & ~31;
    constexpr int KSTEPS = KP / 32;
    constexpr int OHW    = OH * OH;

    __shared__ __align__(16) unsigned short ldsB[16 * KP];

    const int ntile = blockIdx.y;
    const int tid   = threadIdx.x;
    stage_tile_async(Wt + (size_t)ntile * 16 * KP, ldsB, 16 * KP, tid);
    __syncthreads();

    const int wave = tid >> 5, lane = tid & 31;
    const int half = lane >> 4, lm = lane & 15;
    const int mbase0 = blockIdx.x * 128 + wave * 16;   // second tile at +64

    int iy0[2], ix0[2];
    const unsigned short* src[2];
#pragma unroll
    for (int t = 0; t < 2; ++t) {
        int m = mbase0 + t * 64 + lm;
        int bimg = m / OHW;
        int rem  = m - bimg * OHW;
        int oy = rem / OH, ox = rem - (rem / OH) * OH;
        iy0[t] = 2 * oy - 1; ix0[t] = 2 * ox - 1;
        src[t] = actin + (size_t)bimg * (IH * IH * CPIN);
    }

    auto load_chunk = [&](int t, int kb) -> uint4 {
        uint4 zz; zz.x = zz.y = zz.z = zz.w = 0u;
        if (kb >= KREAL) return zz;
        int seg = kb / CPIN;                 // 0..8
        int ci0 = kb - seg * CPIN;           // multiple of 8
        int ky = seg / 3, kx = seg - (seg / 3) * 3;
        int iy = iy0[t] + ky, ix = ix0[t] + kx;
        if ((unsigned)iy < (unsigned)IH && (unsigned)ix < (unsigned)IH)
            return *reinterpret_cast<const uint4*>(&src[t][(iy * IH + ix) * CPIN + ci0]);
        return zz;
    };

    v8f acc0 = {}, acc1 = {};
#pragma unroll
    for (int ks = 0; ks < KSTEPS; ++ks) {
        const int k0 = ks * 32;
        Frag16 a0, a1, b;
        a0.q[0] = load_chunk(0, k0 + half * 8);
        a0.q[1] = load_chunk(0, k0 + 16 + half * 8);
        a1.q[0] = load_chunk(1, k0 + half * 8);
        a1.q[1] = load_chunk(1, k0 + 16 + half * 8);
        const unsigned short* bp = &ldsB[lm * KP + k0 + half * 8];
        b.q[0] = *reinterpret_cast<const uint4*>(bp);
        b.q[1] = *reinterpret_cast<const uint4*>(bp + 16);
        acc0 = __builtin_amdgcn_wmma_f32_16x16x32_bf16(
            false, a0.v, false, b.v, (short)0, acc0, false, false);
        acc1 = __builtin_amdgcn_wmma_f32_16x16x32_bf16(
            false, a1.v, false, b.v, (short)0, acc1, false, false);
    }

    const int co = ntile * 16 + lm;          // always < CPOUT (grid.y = CPOUT/16)
    const bool real = co < COUT;
    const float sc = real ? scale[co] : 0.f;
    const float sh = real ? shift[co] : 0.f;
#pragma unroll
    for (int t = 0; t < 2; ++t) {
        const v8f& acc = t ? acc1 : acc0;
        const int mb = mbase0 + t * 64;
#pragma unroll
        for (int r = 0; r < 8; ++r) {
            int gm = mb + r + 8 * half;
            int b2 = gm / OHW;
            int rm = gm - b2 * OHW;
            int oy2 = rm / OH, ox2 = rm - (rm / OH) * OH;
            float y = 0.f;
            if (real) {
                float g  = gamma[(size_t)b2 * COUT + co];
                float bt = beta [(size_t)b2 * COUT + co];
                y = fmaxf(g * (sc * acc[r] + sh) + bt, 0.f);
            }
            actout[(((size_t)b2 * OH + oy2) * OH + ox2) * CPOUT + co] = f2bf(y);
        }
    }
}

// ---------------------------------------------------------------------------
// LayerNorm(256) + ReLU, fp32 in -> bf16 out. One row per 256-thread block.
// ---------------------------------------------------------------------------
__global__ void ln_relu(const float* __restrict__ h,
                        const float* __restrict__ lnw, const float* __restrict__ lnb,
                        unsigned short* __restrict__ out) {
    const int row = blockIdx.x, tid = threadIdx.x;
    __shared__ float red[256];
    float v = h[(size_t)row * 256 + tid];
    red[tid] = v; __syncthreads();
    for (int s = 128; s > 0; s >>= 1) { if (tid < s) red[tid] += red[tid + s]; __syncthreads(); }
    float mu = red[0] * (1.f / 256.f); __syncthreads();
    float d = v - mu;
    red[tid] = d * d; __syncthreads();
    for (int s = 128; s > 0; s >>= 1) { if (tid < s) red[tid] += red[tid + s]; __syncthreads(); }
    float var = red[0] * (1.f / 256.f);
    float y = d * rsqrtf(var + 1e-5f) * lnw[tid] + lnb[tid];
    out[(size_t)row * 256 + tid] = f2bf(fmaxf(y, 0.f));
}

// out[b] = dot(h2[b], w3) + b3 ; 128 threads per row
__global__ void head3(const unsigned short* __restrict__ h2,
                      const float* __restrict__ w3, const float* __restrict__ b3,
                      float* __restrict__ out) {
    const int row = blockIdx.x, tid = threadIdx.x;
    __shared__ float red[128];
    red[tid] = bf2f(h2[(size_t)row * 128 + tid]) * w3[tid];
    __syncthreads();
    for (int s = 64; s > 0; s >>= 1) { if (tid < s) red[tid] += red[tid + s]; __syncthreads(); }
    if (tid == 0) out[row] = red[0] + b3[0];
}

// ---------------------------------------------------------------------------
// Launcher
// ---------------------------------------------------------------------------
extern "C" void kernel_launch(void* const* d_in, const int* in_sizes, int n_in,
                              void* d_out, int out_size, void* d_ws, size_t ws_size,
                              hipStream_t stream) {
    (void)in_sizes; (void)n_in; (void)out_size; (void)ws_size;

    const float* images = (const float*)d_in[0];
    const int*   z      = (const int*)d_in[1];
    const float* pos    = (const float*)d_in[2];
    const float* embed  = (const float*)d_in[4];
    const float* conv_w[4]; const float* conv_b[4];
    const float* bn_w[4]; const float* bn_b[4]; const float* bn_mean[4]; const float* bn_var[4];
    const float* fgw[4]; const float* fgb[4]; const float* fbw[4]; const float* fbb[4];
    for (int l = 0; l < 4; ++l) {
        conv_w[l]  = (const float*)d_in[5 + l];
        conv_b[l]  = (const float*)d_in[9 + l];
        bn_w[l]    = (const float*)d_in[13 + l];
        bn_b[l]    = (const float*)d_in[17 + l];
        bn_mean[l] = (const float*)d_in[21 + l];
        bn_var[l]  = (const float*)d_in[25 + l];
        fgw[l]     = (const float*)d_in[29 + l];
        fgb[l]     = (const float*)d_in[33 + l];
        fbw[l]     = (const float*)d_in[37 + l];
        fbb[l]     = (const float*)d_in[41 + l];
    }
    const float* head_w1 = (const float*)d_in[45];
    const float* head_b1 = (const float*)d_in[46];
    const float* ln_w    = (const float*)d_in[47];
    const float* ln_b    = (const float*)d_in[48];
    const float* head_w2 = (const float*)d_in[49];
    const float* head_b2 = (const float*)d_in[50];
    const float* head_w3 = (const float*)d_in[51];
    const float* head_b3 = (const float*)d_in[52];
    float* out = (float*)d_out;

    static const int C[4]   = {48, 96, 104, 52};   // real out channels
    static const int CPI[4] = {16, 48, 96, 112};   // padded in channels
    static const int CPO[4] = {48, 96, 112, 64};   // padded out channels (=NP)
    static const int CIR[4] = {10, 48, 96, 104};   // real in channels
    static const int KPc[4] = {160, 448, 864, 1024};

    // ---- workspace carve-out ----
    size_t off = 0;
    auto carve = [&](size_t bytes) -> char* {
        char* p = (char*)d_ws + off;
        off = (off + bytes + 255) & ~(size_t)255;
        return p;
    };
    unsigned short* act0 = (unsigned short*)carve((size_t)BATCH * 1024 * 16 * 2);
    unsigned short* act1 = (unsigned short*)carve((size_t)BATCH * 256 * 48 * 2);
    unsigned short* act2 = (unsigned short*)carve((size_t)BATCH * 64 * 96 * 2);
    unsigned short* act3 = (unsigned short*)carve((size_t)BATCH * 16 * 112 * 2);
    unsigned short* act4 = (unsigned short*)carve((size_t)BATCH * 4 * 64 * 2);
    unsigned short* cond = (unsigned short*)carve((size_t)BATCH * CONDP * 2);
    unsigned short* wp[4]; unsigned short* gwp[4]; unsigned short* bwp[4];
    float *scl[4], *shf[4], *gam[4], *bet[4];
    for (int l = 0; l < 4; ++l) {
        wp[l]  = (unsigned short*)carve((size_t)CPO[l] * KPc[l] * 2);
        gwp[l] = (unsigned short*)carve((size_t)CPO[l] * CONDP * 2);
        bwp[l] = (unsigned short*)carve((size_t)CPO[l] * CONDP * 2);
        scl[l] = (float*)carve((size_t)C[l] * 4);
        shf[l] = (float*)carve((size_t)C[l] * 4);
        gam[l] = (float*)carve((size_t)BATCH * C[l] * 4);
        bet[l] = (float*)carve((size_t)BATCH * C[l] * 4);
    }
    unsigned short* w1p = (unsigned short*)carve((size_t)256 * 256 * 2);
    unsigned short* w2p = (unsigned short*)carve((size_t)128 * 256 * 2);
    float*          h1  = (float*)carve((size_t)BATCH * 256 * 4);
    unsigned short* h1b = (unsigned short*)carve((size_t)BATCH * 256 * 2);
    unsigned short* h2b = (unsigned short*)carve((size_t)BATCH * 128 * 2);

    // ---- prep ----
    {
        int n = BATCH * 1024 * 16;
        img_to_nhwc<<<(n + 255) / 256, 256, 0, stream>>>(images, act0);
    }
    build_cond<<<(BATCH * CONDP + 255) / 256, 256, 0, stream>>>(z, pos, embed, cond);
    for (int l = 0; l < 4; ++l) {
        fuse_bn<<<1, 128, 0, stream>>>(bn_w[l], bn_b[l], bn_mean[l], bn_var[l],
                                       conv_b[l], scl[l], shf[l], C[l]);
        int ne = CPO[l] * KPc[l];
        pack_conv<<<(ne + 255) / 256, 256, 0, stream>>>(conv_w[l], wp[l],
                                                        C[l], CIR[l], CPI[l], CPO[l], KPc[l]);
        ne = CPO[l] * CONDP;
        pack_fc<<<(ne + 255) / 256, 256, 0, stream>>>(fgw[l], gwp[l], C[l], 113, CPO[l], CONDP);
        pack_fc<<<(ne + 255) / 256, 256, 0, stream>>>(fbw[l], bwp[l], C[l], 113, CPO[l], CONDP);
    }
    pack_head1<<<(256 * 256 + 255) / 256, 256, 0, stream>>>(head_w1, w1p);
    pack_fc<<<(128 * 256 + 255) / 256, 256, 0, stream>>>(head_w2, w2p, 128, 256, 128, 256);

    // ---- FiLM gamma/beta: [4096 x 128] @ [128 x C] per layer (WMMA) ----
    for (int l = 0; l < 4; ++l) {
        dim3 g(BATCH / 128, CPO[l] / 16);
        gemm_wmma<CONDP, false, false><<<g, 128, 0, stream>>>(
            cond, gwp[l], fgb[l], gam[l], BATCH, C[l], C[l]);
        gemm_wmma<CONDP, false, false><<<g, 128, 0, stream>>>(
            cond, bwp[l], fbb[l], bet[l], BATCH, C[l], C[l]);
    }

    // ---- conv stack (implicit GEMM, WMMA, NHWC, dual M-tiles/wave) ----
    conv_film_wmma<10, 16, 48, 48, 32, 16><<<dim3(BATCH * 256 / 128, 3), 128, 0, stream>>>(
        act0, wp[0], scl[0], shf[0], gam[0], bet[0], act1);
    conv_film_wmma<48, 48, 96, 96, 16, 8><<<dim3(BATCH * 64 / 128, 6), 128, 0, stream>>>(
        act1, wp[1], scl[1], shf[1], gam[1], bet[1], act2);
    conv_film_wmma<96, 96, 104, 112, 8, 4><<<dim3(BATCH * 16 / 128, 7), 128, 0, stream>>>(
        act2, wp[2], scl[2], shf[2], gam[2], bet[2], act3);
    conv_film_wmma<104, 112, 52, 64, 4, 2><<<dim3(BATCH * 4 / 128, 4), 128, 0, stream>>>(
        act3, wp[3], scl[3], shf[3], gam[3], bet[3], act4);

    // ---- head (act4 flat [B][256] == padded flatten; permutation absorbed in w1p) ----
    gemm_wmma<256, false, false><<<dim3(BATCH / 128, 16), 128, 0, stream>>>(
        act4, w1p, head_b1, h1, BATCH, 256, 256);
    ln_relu<<<BATCH, 256, 0, stream>>>(h1, ln_w, ln_b, h1b);
    gemm_wmma<256, true, true><<<dim3(BATCH / 128, 8), 128, 0, stream>>>(
        h1b, w2p, head_b2, h2b, BATCH, 128, 128);
    head3<<<BATCH, 128, 0, stream>>>(h2b, head_w3, head_b3, out);
}
